// CrissCrossAttention_58282706207079
// MI455X (gfx1250) — matrix operand order repaired and checked by reading, not compiled, not run on hardware
//
#include <hip/hip_runtime.h>
#include <hip/hip_bf16.h>

// ---------------------------------------------------------------------------
// CrissCross attention for x:(B=8,H=128,W=128,C=64), d=C/8=8.
// bf16 WMMA (v_wmma_f32_16x16x32_bf16) for all GEMMs, fp32 softmax.
// Pass H (per b,w column, diag-masked) -> unnormalized oH + stats(m,s).
// Pass W (per b,h row) fuses the exact two-way softmax merge + residual
// and writes the final output, eliminating the oW buffer and a 4th kernel.
// Q/K LDS staging uses the CDNA5 async global->LDS DMA path (ASYNCcnt).
// ---------------------------------------------------------------------------

#define BB 8
#define HH 128
#define WW 128
#define CC 64
#define DD 8
#define NPOS (BB*HH*WW)          // 131072

typedef __bf16 v16bf __attribute__((ext_vector_type(16)));
typedef float  v8f   __attribute__((ext_vector_type(8)));

union BOp {                      // one WMMA A/B operand (16 bf16 = 8 VGPRs)
    v16bf v;
    uint4 u[2];
    unsigned short s[16];
};

__device__ __forceinline__ unsigned short f2bf(float f) {
    __bf16 h = (__bf16)f;
    return __builtin_bit_cast(unsigned short, h);
}

__device__ __forceinline__ v8f wmma_bf16(const BOp& a, const BOp& b, v8f c) {
    return __builtin_amdgcn_wmma_f32_16x16x32_bf16(
        /*neg_a=*/false, a.v, /*neg_b=*/false, b.v,
        /*c_mod=*/(short)0, c, /*reuse_a=*/false, /*reuse_b=*/false);
}

#define ZERO8 {0.f,0.f,0.f,0.f,0.f,0.f,0.f,0.f}

// ---------------------------------------------------------------------------
// Kernel 1: projections.  Q|K fused as one N=16 tile, V = 4 N-tiles.
// Weights staged transposed (N-major) in LDS so B-operands are contiguous.
// ---------------------------------------------------------------------------
__global__ __launch_bounds__(256) void cc_proj_kernel(
    const float* __restrict__ x,
    const float* __restrict__ wq, const float* __restrict__ bq,
    const float* __restrict__ wk, const float* __restrict__ bk,
    const float* __restrict__ wv, const float* __restrict__ bv,
    unsigned short* __restrict__ qh, unsigned short* __restrict__ kh,
    unsigned short* __restrict__ vh)
{
    __shared__ __align__(16) unsigned short wqkT[16][64];   // [n][k]
    __shared__ __align__(16) unsigned short wvT [64][64];   // [n][k]
    __shared__ float biasqk[16];
    __shared__ float biasv [64];

    const int tid = threadIdx.x;

    for (int i = tid; i < 16*64; i += 256) {
        int n = i >> 6, k = i & 63;
        float v = (n < 8) ? wq[k*8 + n] : wk[k*8 + (n - 8)];
        wqkT[n][k] = f2bf(v);
    }
    for (int i = tid; i < 64*64; i += 256) {
        int n = i >> 6, k = i & 63;
        wvT[n][k] = f2bf(wv[k*64 + n]);
    }
    if (tid < 16) biasqk[tid] = (tid < 8) ? bq[tid] : bk[tid - 8];
    if (tid < 64) biasv[tid]  = bv[tid];
    __syncthreads();

    const int lane = tid & 31, wave = tid >> 5;
    const int ml   = lane & 15;            // M row (A) / N col (B,C)
    const int hsel = lane >> 4;            // lane-half select
    const int tile = blockIdx.x * 8 + wave;    // 8192 tiles of 16 rows
    const int row0 = tile * 16;

    // A operands: x rows, fp32 -> bf16 on the fly.  K chunks per wave32 layout.
    BOp a[2];
    {
        const float* xr = x + (size_t)(row0 + ml) * CC;
        #pragma unroll
        for (int ks = 0; ks < 2; ++ks) {
            const float* p0 = xr + 32*ks + 16*hsel;      // K = 32ks+16h .. +7
            const float* p1 = p0 + 8;                    // K = +8 .. +15
            #pragma unroll
            for (int e = 0; e < 8; ++e) {
                a[ks].s[e]     = f2bf(p0[e]);
                a[ks].s[8 + e] = f2bf(p1[e]);
            }
        }
    }

    v8f accQK = ZERO8;
    v8f accV[4] = { ZERO8, ZERO8, ZERO8, ZERO8 };

    #pragma unroll
    for (int ks = 0; ks < 2; ++ks) {
        BOp b;
        const unsigned short* pw = &wqkT[ml][32*ks + 16*hsel];
        b.u[0] = *(const uint4*)pw;
        b.u[1] = *(const uint4*)(pw + 8);
        accQK = wmma_bf16(a[ks], b, accQK);
        #pragma unroll
        for (int nt = 0; nt < 4; ++nt) {
            const unsigned short* pv = &wvT[nt*16 + ml][32*ks + 16*hsel];
            b.u[0] = *(const uint4*)pv;
            b.u[1] = *(const uint4*)(pv + 8);
            accV[nt] = wmma_bf16(a[ks], b, accV[nt]);
        }
    }

    // C layout: VGPR j -> rows j (lanes 0-15) / j+8 (lanes 16-31), col = ml.
    const float bqk_l = biasqk[ml];
    #pragma unroll
    for (int j = 0; j < 8; ++j) {
        int r = row0 + j + 8*hsel;
        float qv = accQK[j] + bqk_l;
        if (ml < 8) qh[r*DD + ml]       = f2bf(qv);
        else        kh[r*DD + (ml - 8)] = f2bf(qv);
        #pragma unroll
        for (int nt = 0; nt < 4; ++nt)
            vh[r*CC + nt*16 + ml] = f2bf(accV[nt][j] + biasv[nt*16 + ml]);
    }
}

// ---------------------------------------------------------------------------
// Kernel 2: one attention axis.
//  dir=0: per-(b,w) column over h, diag mask; writes unnormalized oH + stats.
//  dir=1: per-(b,h) row over w; fuses exact softmax merge with the H-pass
//         partials and the residual, writing the final output.
// ---------------------------------------------------------------------------
__global__ __launch_bounds__(256) void cc_attn_kernel(
    const unsigned short* __restrict__ qh,
    const unsigned short* __restrict__ kh,
    const unsigned short* __restrict__ vh,
    float* __restrict__ oH, float2* __restrict__ stH, int dir,
    const float* __restrict__ xg, const float* __restrict__ gamma,
    float* __restrict__ finalOut)
{
    __shared__ __align__(16) unsigned short Qs[128][8];
    __shared__ __align__(16) unsigned short Ks[128][8];
    __shared__ __align__(16) unsigned short Vt[64][128];       // transposed V
    __shared__ __align__(16) unsigned short Ps[8][16][128];    // per-wave P

    const int tid = threadIdx.x, lane = tid & 31, wave = tid >> 5;
    const int ml = lane & 15, hsel = lane >> 4;
    const int blk = blockIdx.x;                 // b*128 + fixed-coord
    const int bb = blk >> 7, f = blk & 127;
    const size_t posBase   = dir ? (size_t)(bb*128 + f) * 128
                                 : (size_t)bb * 128 * 128 + f;
    const int    posStride = dir ? 1 : 128;

    // Stage Q,K (128x8 each) via async global->LDS DMA (no VGPR round-trip).
    {
        int p = tid & 127;
        size_t idx = (posBase + (size_t)p * posStride) * DD;
        const unsigned short* src = (tid < 128) ? (qh + idx) : (kh + idx);
        unsigned short* dst = (tid < 128) ? &Qs[p][0] : &Ks[p][0];
        unsigned int lds_off = (unsigned int)(size_t)dst;   // low 32b = LDS addr
        asm volatile("global_load_async_to_lds_b128 %0, %1, off"
                     :: "v"(lds_off), "v"(src) : "memory");
    }
    // Stage V transposed (Vt[c][p]) through VGPRs.
    for (int i = tid; i < 128*8; i += 256) {
        int p = i >> 3, cg = (i & 7) * 8;
        uint4 d = *(const uint4*)(vh + (posBase + (size_t)p * posStride)*CC + cg);
        const unsigned short* e = (const unsigned short*)&d;
        #pragma unroll
        for (int j = 0; j < 8; ++j) Vt[cg + j][p] = e[j];
    }
    asm volatile("s_wait_asynccnt 0x0" ::: "memory");
    __syncthreads();

    const int m0 = wave * 16;                   // query rows for this wave

    // A operand: Q rows, d=8 zero-padded to K=32 (upper chunks/lanes zero).
    BOp aQ;
    aQ.u[0] = make_uint4(0,0,0,0); aQ.u[1] = make_uint4(0,0,0,0);
    if (!hsel) aQ.u[0] = *(const uint4*)&Qs[m0 + ml][0];

    // Energies: 8 key tiles of 16.
    v8f E[8];
    #pragma unroll
    for (int t = 0; t < 8; ++t) {
        BOp bK;
        bK.u[0] = make_uint4(0,0,0,0); bK.u[1] = make_uint4(0,0,0,0);
        if (!hsel) bK.u[0] = *(const uint4*)&Ks[t*16 + ml][0];
        v8f c = ZERO8;
        E[t] = wmma_bf16(aQ, bK, c);
    }

    // Diagonal mask for the H pass (query h == key k).
    if (dir == 0) {
        #pragma unroll
        for (int t = 0; t < 8; ++t)
            #pragma unroll
            for (int j = 0; j < 8; ++j)
                if (t*16 + ml == m0 + j + 8*hsel) E[t][j] = -3.0e38f;
    }

    // Row-wise max / exp / sum (rows live entirely in one 16-lane half;
    // butterfly reduction leaves the result in every lane of the half).
    float mrow[8], srow[8];
    #pragma unroll
    for (int j = 0; j < 8; ++j) {
        float v = E[0][j];
        #pragma unroll
        for (int t = 1; t < 8; ++t) v = fmaxf(v, E[t][j]);
        #pragma unroll
        for (int off = 1; off < 16; off <<= 1)
            v = fmaxf(v, __shfl_xor(v, off, 32));
        mrow[j] = v;
        float s = 0.f;
        #pragma unroll
        for (int t = 0; t < 8; ++t) {
            float p = __expf(E[t][j] - v);
            E[t][j] = p;
            s += p;
        }
        #pragma unroll
        for (int off = 1; off < 16; off <<= 1)
            s += __shfl_xor(s, off, 32);
        srow[j] = s;
    }

    if (dir == 0 && (lane & 15) == 0) {
        #pragma unroll
        for (int j = 0; j < 8; ++j)
            stH[blk*128 + m0 + j + 8*hsel] = make_float2(mrow[j], srow[j]);
    }

    // C-layout P -> row-major LDS tile (for A-layout reload).
    #pragma unroll
    for (int t = 0; t < 8; ++t)
        #pragma unroll
        for (int j = 0; j < 8; ++j)
            Ps[wave][j + 8*hsel][t*16 + ml] = f2bf(E[t][j]);

    // out = P (16x128) x V (128x64): 4 K-steps x 4 N-tiles.
    v8f O[4] = { ZERO8, ZERO8, ZERO8, ZERO8 };
    #pragma unroll
    for (int ks = 0; ks < 4; ++ks) {
        BOp aP;
        const unsigned short* pr = &Ps[wave][ml][32*ks + 16*hsel];
        aP.u[0] = *(const uint4*)pr;
        aP.u[1] = *(const uint4*)(pr + 8);
        #pragma unroll
        for (int nt = 0; nt < 4; ++nt) {
            BOp bV;
            const unsigned short* vr = &Vt[nt*16 + ml][32*ks + 16*hsel];
            bV.u[0] = *(const uint4*)vr;
            bV.u[1] = *(const uint4*)(vr + 8);
            O[nt] = wmma_bf16(aP, bV, O[nt]);
        }
    }

    if (dir == 0) {
        // H pass: store unnormalized partial sums.
        #pragma unroll
        for (int nt = 0; nt < 4; ++nt)
            #pragma unroll
            for (int j = 0; j < 8; ++j) {
                int p = m0 + j + 8*hsel;
                size_t oidx = (posBase + (size_t)p * posStride)*CC + nt*16 + ml;
                oH[oidx] = O[nt][j];
            }
    } else {
        // W pass: exact merge of both partial softmaxes + residual.
        const float gm = gamma[0];
        #pragma unroll
        for (int j = 0; j < 8; ++j) {
            int p = m0 + j + 8*hsel;              // query w
            float2 sH = stH[(bb*128 + p)*128 + f];
            float  mW = mrow[j], sW = srow[j];
            float  M  = fmaxf(sH.x, mW);
            float  al = __expf(sH.x - M);
            float  be = __expf(mW - M);
            float  g  = gm / (sH.y * al + sW * be);
            float  sA = g * al, sB = g * be;
            #pragma unroll
            for (int nt = 0; nt < 4; ++nt) {
                size_t oidx = (posBase + (size_t)p)*CC + nt*16 + ml;
                finalOut[oidx] = oH[oidx]*sA + O[nt][j]*sB + xg[oidx];
            }
        }
    }
}

// ---------------------------------------------------------------------------
extern "C" void kernel_launch(void* const* d_in, const int* in_sizes, int n_in,
                              void* d_out, int out_size, void* d_ws, size_t ws_size,
                              hipStream_t stream) {
    const float* x  = (const float*)d_in[0];
    const float* wq = (const float*)d_in[1];
    const float* bq = (const float*)d_in[2];
    const float* wk = (const float*)d_in[3];
    const float* bk = (const float*)d_in[4];
    const float* wv = (const float*)d_in[5];
    const float* bv = (const float*)d_in[6];
    const float* gamma = (const float*)d_in[7];
    float* out = (float*)d_out;

    // Workspace layout (bf16 stored as ushort).
    unsigned short* qh = (unsigned short*)d_ws;                 // NPOS*8
    unsigned short* kh = qh + (size_t)NPOS * DD;                // NPOS*8
    unsigned short* vh = kh + (size_t)NPOS * DD;                // NPOS*64
    float*  oH  = (float*)(vh + (size_t)NPOS * CC);             // NPOS*64 f32
    float2* stH = (float2*)(oH + (size_t)NPOS * CC);            // NPOS

    cc_proj_kernel<<<1024, 256, 0, stream>>>(x, wq, bq, wk, bk, wv, bv,
                                             qh, kh, vh);
    cc_attn_kernel<<<BB*WW, 256, 0, stream>>>(qh, kh, vh, oH, stH, /*dir=*/0,
                                              x, gamma, out);
    cc_attn_kernel<<<BB*HH, 256, 0, stream>>>(qh, kh, vh, oH, stH, /*dir=*/1,
                                              x, gamma, out);
}
